// MultiheadAttentionSDPAWithRoPE_28484223107508
// MI455X (gfx1250) — compile-verified, matrix operand order checked by reading, not run on hardware
//
#include <hip/hip_runtime.h>
#include <hip/hip_bf16.h>
#include <math.h>

typedef _Float16 h16;
typedef __attribute__((ext_vector_type(16))) _Float16 v16h;
typedef __attribute__((ext_vector_type(8)))  _Float16 v8h;
typedef __attribute__((ext_vector_type(8)))  float    v8f;

#define BB 2
#define TT 2048
#define DD 1024
#define HH 16
#define HD 64
#define EPSF 1.1920928955078125e-7f

// Load a 16-row x 32-K f16 WMMA fragment (A or B operand) from a row-major
// matrix. lane = row (M for A, N for B); per-lane K set:
//   K = (e&7) + 16*(e>>3) + 8*(lane>=16)   (CDNA5 16-bit 16x32 layout)
// -> two contiguous 16B loads per lane.
__device__ __forceinline__ v16h load_frag(const h16* __restrict__ base, int ld, int k0) {
  const int lane = threadIdx.x & 31;
  const int hi = lane >> 4;
  const h16* p = base + (size_t)(lane & 15) * ld + k0 + hi * 8;
  v8h a = *(const v8h*)p;        // K = k0 + hi*8 + 0..7        -> e 0..7
  v8h b = *(const v8h*)(p + 16); // K = k0 + 16 + hi*8 + 0..7   -> e 8..15
  return __builtin_shufflevector(a, b, 0,1,2,3,4,5,6,7,8,9,10,11,12,13,14,15);
}

// ---------------- generic f16 WMMA GEMM: C[M][N] = A[M][K] * Bt[N][K]^T + bias ----
// Software-pipelined with a peeled tail: loop offsets stay affine (immediate
// offsets / pointer bumps), WMMAs only wait on loads issued one iteration back.
__global__ void __launch_bounds__(256)
gemm16(const h16* __restrict__ A, const h16* __restrict__ Bt,
       const float* __restrict__ bias, float* __restrict__ C,
       int M, int N, int K) {
  const int wave = threadIdx.x >> 5;
  const int lane = threadIdx.x & 31;
  const int m0 = blockIdx.x * 64 + (wave & 3) * 16;   // 4 waves along M
  const int n0 = blockIdx.y * 128 + (wave >> 2) * 64; // 2 waves along N, 4 n-tiles each
  v8f acc[4] = {};
  const h16* Arow = A + (size_t)m0 * K;
  const h16* B0 = Bt + (size_t)(n0 + 0) * K;
  const h16* B1 = Bt + (size_t)(n0 + 16) * K;
  const h16* B2 = Bt + (size_t)(n0 + 32) * K;
  const h16* B3 = Bt + (size_t)(n0 + 48) * K;

  v16h af = load_frag(Arow, K, 0);
  v16h b0 = load_frag(B0, K, 0);
  v16h b1 = load_frag(B1, K, 0);
  v16h b2 = load_frag(B2, K, 0);
  v16h b3 = load_frag(B3, K, 0);

  const h16* Apf = Arow + (size_t)(lane & 15) * K;
  for (int k = 0; k + 32 < K; k += 32) {
    // WGP-scope prefetch two K-steps ahead (warms L0; harmless at the edge)
    __builtin_prefetch(Apf + k + 64, 0, 3);
    // issue next-iteration fragment loads first (pipelining, affine offsets)
    v16h afn = load_frag(Arow, K, k + 32);
    v16h b0n = load_frag(B0, K, k + 32);
    v16h b1n = load_frag(B1, K, k + 32);
    v16h b2n = load_frag(B2, K, k + 32);
    v16h b3n = load_frag(B3, K, k + 32);
    // consume current fragments
    acc[0] = __builtin_amdgcn_wmma_f32_16x16x32_f16(false, af, false, b0, (short)0, acc[0], false, false);
    acc[1] = __builtin_amdgcn_wmma_f32_16x16x32_f16(false, af, false, b1, (short)0, acc[1], false, false);
    acc[2] = __builtin_amdgcn_wmma_f32_16x16x32_f16(false, af, false, b2, (short)0, acc[2], false, false);
    acc[3] = __builtin_amdgcn_wmma_f32_16x16x32_f16(false, af, false, b3, (short)0, acc[3], false, false);
    af = afn; b0 = b0n; b1 = b1n; b2 = b2n; b3 = b3n;
  }
  // peeled final K-step (no loads)
  acc[0] = __builtin_amdgcn_wmma_f32_16x16x32_f16(false, af, false, b0, (short)0, acc[0], false, false);
  acc[1] = __builtin_amdgcn_wmma_f32_16x16x32_f16(false, af, false, b1, (short)0, acc[1], false, false);
  acc[2] = __builtin_amdgcn_wmma_f32_16x16x32_f16(false, af, false, b2, (short)0, acc[2], false, false);
  acc[3] = __builtin_amdgcn_wmma_f32_16x16x32_f16(false, af, false, b3, (short)0, acc[3], false, false);

  const int col = lane & 15;
  const int hi8 = (lane >> 4) * 8;
#pragma unroll
  for (int j = 0; j < 4; ++j) {
    const int n = n0 + j * 16 + col;
    const float bv = bias ? bias[n] : 0.f;
#pragma unroll
    for (int r = 0; r < 8; ++r)
      C[(size_t)(m0 + r + hi8) * N + n] = acc[j][r] + bv;
  }
}

// ---------------- elementwise converts / transposes --------------------------------
__global__ void cvt_f32_f16(const float* __restrict__ in, h16* __restrict__ out, int n) {
  int i = blockIdx.x * 256 + threadIdx.x;
  if (i < n) out[i] = (h16)in[i];
}

// in: [K][N] row-major f32  ->  out: [N][K] f16 (K-major for WMMA B operands)
__global__ void transpose_f32_f16(const float* __restrict__ in, h16* __restrict__ out,
                                  int K, int N) {
  int i = blockIdx.x * 256 + threadIdx.x;
  if (i >= N * K) return;
  int k = i % K, n = i / K;
  out[i] = (h16)in[(size_t)k * N + n];
}

// v f32 [B*T][D] (d = h*64+hd)  ->  Vt f16 [B][H][HD][T]
__global__ void transposeV(const float* __restrict__ v, h16* __restrict__ Vt) {
  size_t i = (size_t)blockIdx.x * 256 + threadIdx.x; // B*H*HD*T = 4,194,304
  int t = (int)(i & (TT - 1));
  size_t r = i >> 11;
  int hd = (int)(r & (HD - 1));
  size_t bh = r >> 6; // b*16+h
  int h = (int)(bh & 15), b = (int)(bh >> 4);
  Vt[i] = (h16)(v[((size_t)(b * TT + t)) * DD + h * HD + hd]);
}

// ---------------- fused RMSNorm + RoPE, f32 qk rows -> f16 [B][H][T][HD] ----------
__global__ void __launch_bounds__(256)
normrope(const float* __restrict__ qk, const float* __restrict__ qw,
         const float* __restrict__ kw, const float* __restrict__ cosT,
         const float* __restrict__ sinT, h16* __restrict__ Qo, h16* __restrict__ Ko) {
  const int t = blockIdx.x, b = blockIdx.y;
  const int tid = threadIdx.x;
  __shared__ float sq[DD], sk[DD], red[18];
  const float* qrow = qk + (size_t)(b * TT + t) * (2 * DD);
  const float* krow = qrow + DD;
  float aq = 0.f, ak = 0.f;
  for (int i = tid; i < DD; i += 256) {
    float q = qrow[i], k = krow[i];
    sq[i] = q; sk[i] = k;
    aq += q * q; ak += k * k;
  }
#pragma unroll
  for (int off = 16; off > 0; off >>= 1) {
    aq += __shfl_xor(aq, off, 32);
    ak += __shfl_xor(ak, off, 32);
  }
  if ((tid & 31) == 0) { red[tid >> 5] = aq; red[(tid >> 5) + 8] = ak; }
  __syncthreads();
  if (tid == 0) {
    float a = 0.f, c = 0.f;
    for (int w = 0; w < 8; ++w) { a += red[w]; c += red[w + 8]; }
    red[16] = rsqrtf(a / DD + EPSF);
    red[17] = rsqrtf(c / DD + EPSF);
  }
  __syncthreads();
  const float rq = red[16], rk = red[17];
  for (int p = tid; p < DD / 2; p += 256) {
    const int h = p >> 5;    // 32 rotation pairs per head
    const int i = p & 31;
    const int d = h * HD + 2 * i;
    const float c = cosT[t * 32 + i], s = sinT[t * 32 + i];
    const float qa = sq[d] * rq * qw[d],     qb = sq[d + 1] * rq * qw[d + 1];
    const float ka = sk[d] * rk * kw[d],     kb = sk[d + 1] * rk * kw[d + 1];
    const size_t o = ((size_t)(b * HH + h) * TT + t) * HD + 2 * i;
    Qo[o]     = (h16)(qa * c - qb * s);
    Qo[o + 1] = (h16)(qa * s + qb * c);
    Ko[o]     = (h16)(ka * c - kb * s);
    Ko[o + 1] = (h16)(ka * s + kb * c);
  }
}

// ---------------- causal flash attention: one wave = 16 queries --------------------
// S^T = K * Q^T  (lane = query, vgpr = key) -> per-lane softmax + 1 shfl.
// O^T = V^T * P^T; the exp'd S^T regs of two 16-key tiles ARE the P B-fragment.
// K fragments are pipelined one iteration ahead; V fragments load before the
// softmax VALU block, so WMMAs never wait on a cold load.
__global__ void __launch_bounds__(128)
attn_fwd(const h16* __restrict__ Q, const h16* __restrict__ Km,
         const h16* __restrict__ Vt, h16* __restrict__ O) {
  const int wave = threadIdx.x >> 5;
  const int lane = threadIdx.x & 31;
  const int q0 = (blockIdx.x * 4 + wave) * 16;
  const int h = blockIdx.y, b = blockIdx.z;
  const size_t bh = (size_t)b * HH + h;
  const h16* Qh = Q + bh * TT * HD;
  const h16* Kh = Km + bh * TT * HD;
  const h16* Vh = Vt + bh * (size_t)HD * TT;

  const v16h qf0 = load_frag(Qh + (size_t)q0 * HD, HD, 0);
  const v16h qf1 = load_frag(Qh + (size_t)q0 * HD, HD, 32);

  // prologue: K fragments for kb = 0
  v16h ka0 = load_frag(Kh, HD, 0);
  v16h ka1 = load_frag(Kh, HD, 32);
  v16h ka2 = load_frag(Kh + (size_t)16 * HD, HD, 0);
  v16h ka3 = load_frag(Kh + (size_t)16 * HD, HD, 32);

  v8f oacc[4] = {};
  float m = -INFINITY, lsum = 0.f;
  const int qrow = q0 + (lane & 15);
  const int khi = (lane >> 4) * 8;
  const int kend = q0 + 16;

  for (int kb = 0; kb < kend; kb += 32) {
    v8f s0 = {}, s1 = {};
    s0 = __builtin_amdgcn_wmma_f32_16x16x32_f16(false, ka0, false, qf0, (short)0, s0, false, false);
    s0 = __builtin_amdgcn_wmma_f32_16x16x32_f16(false, ka1, false, qf1, (short)0, s0, false, false);
    s1 = __builtin_amdgcn_wmma_f32_16x16x32_f16(false, ka2, false, qf0, (short)0, s1, false, false);
    s1 = __builtin_amdgcn_wmma_f32_16x16x32_f16(false, ka3, false, qf1, (short)0, s1, false, false);

    // pipeline: issue next iteration's K fragments (dummy kb=0 on last iter)
    const int kn = (kb + 32 < kend) ? (kb + 32) : 0;
    v16h kn0 = load_frag(Kh + (size_t)kn * HD, HD, 0);
    v16h kn1 = load_frag(Kh + (size_t)kn * HD, HD, 32);
    v16h kn2 = load_frag(Kh + (size_t)(kn + 16) * HD, HD, 0);
    v16h kn3 = load_frag(Kh + (size_t)(kn + 16) * HD, HD, 32);
    // current V^T fragments; softmax VALU below hides their latency
    v16h va0 = load_frag(Vh + (size_t)(0 * 16) * TT, TT, kb);
    v16h va1 = load_frag(Vh + (size_t)(1 * 16) * TT, TT, kb);
    v16h va2 = load_frag(Vh + (size_t)(2 * 16) * TT, TT, kb);
    v16h va3 = load_frag(Vh + (size_t)(3 * 16) * TT, TT, kb);

    float sv0[8], sv1[8];
    float tmax = -INFINITY;
#pragma unroll
    for (int r = 0; r < 8; ++r) {
      const int key = kb + khi + r;
      sv0[r] = (key <= qrow)      ? s0[r] * 0.125f : -INFINITY; // 1/sqrt(64)
      sv1[r] = (key + 16 <= qrow) ? s1[r] * 0.125f : -INFINITY;
      tmax = fmaxf(tmax, fmaxf(sv0[r], sv1[r]));
    }
    tmax = fmaxf(tmax, __shfl_xor(tmax, 16, 32));
    const float mnew = fmaxf(m, tmax);
    const float alpha = __expf(m - mnew); // first iter: exp(-inf)=0
    m = mnew;
    float psum = 0.f;
    v16h pf;
#pragma unroll
    for (int r = 0; r < 8; ++r) {
      const float p0 = __expf(sv0[r] - mnew);
      const float p1 = __expf(sv1[r] - mnew);
      psum += p0 + p1;
      pf[r] = (h16)p0;      // B-frag e<8  : keys kb+khi+r
      pf[r + 8] = (h16)p1;  // B-frag e>=8 : keys kb+16+khi+r
    }
    psum += __shfl_xor(psum, 16, 32);
    lsum = lsum * alpha + psum;
#pragma unroll
    for (int j = 0; j < 4; ++j)
#pragma unroll
      for (int r = 0; r < 8; ++r) oacc[j][r] *= alpha;

    oacc[0] = __builtin_amdgcn_wmma_f32_16x16x32_f16(false, va0, false, pf, (short)0, oacc[0], false, false);
    oacc[1] = __builtin_amdgcn_wmma_f32_16x16x32_f16(false, va1, false, pf, (short)0, oacc[1], false, false);
    oacc[2] = __builtin_amdgcn_wmma_f32_16x16x32_f16(false, va2, false, pf, (short)0, oacc[2], false, false);
    oacc[3] = __builtin_amdgcn_wmma_f32_16x16x32_f16(false, va3, false, pf, (short)0, oacc[3], false, false);

    ka0 = kn0; ka1 = kn1; ka2 = kn2; ka3 = kn3;
  }
  const float inv = 1.f / lsum;
  h16* orow = O + ((size_t)b * TT + qrow) * DD + h * HD + khi;
#pragma unroll
  for (int j = 0; j < 4; ++j) {
    v8h ov;
#pragma unroll
    for (int r = 0; r < 8; ++r) ov[r] = (h16)(oacc[j][r] * inv);
    *(v8h*)(orow + j * 16) = ov;
  }
}

// ---------------- host launch ------------------------------------------------------
extern "C" void kernel_launch(void* const* d_in, const int* in_sizes, int n_in,
                              void* d_out, int out_size, void* d_ws, size_t ws_size,
                              hipStream_t stream) {
  const float* x   = (const float*)d_in[0];
  const float* fc  = (const float*)d_in[1];
  const float* fs  = (const float*)d_in[2];
  const float* Wqk = (const float*)d_in[3];
  const float* bqk = (const float*)d_in[4];
  const float* Wv  = (const float*)d_in[5];
  const float* Wo  = (const float*)d_in[6];
  const float* qw  = (const float*)d_in[7];
  const float* kw  = (const float*)d_in[8];
  float* out = (float*)d_out;
  (void)in_sizes; (void)n_in; (void)out_size; (void)ws_size;

  char* ws = (char*)d_ws;
  size_t off = 0;
  auto alloc = [&](size_t bytes) {
    void* p = ws + off;
    off += (bytes + 255) & ~(size_t)255;
    return p;
  };
  h16*   Xf   = (h16*)alloc((size_t)4096 * 1024 * 2);   // x in f16
  h16*   Wqkt = (h16*)alloc((size_t)2048 * 1024 * 2);   // Wqk^T f16 [N][K]
  h16*   Wvt  = (h16*)alloc((size_t)1024 * 1024 * 2);
  h16*   Wot  = (h16*)alloc((size_t)1024 * 1024 * 2);
  float* qkb  = (float*)alloc((size_t)4096 * 2048 * 4); // qk projection f32
  float* vb   = (float*)alloc((size_t)4096 * 1024 * 4); // v projection f32
  h16*   Q16  = (h16*)alloc((size_t)BB * HH * TT * HD * 2);
  h16*   K16  = (h16*)alloc((size_t)BB * HH * TT * HD * 2);
  h16*   Vt16 = Xf;        // alias: Xf dead after v GEMM
  h16*   A16  = (h16*)qkb; // alias: qkb dead after normrope

  cvt_f32_f16<<<16384, 256, 0, stream>>>(x, Xf, 4194304);
  transpose_f32_f16<<<8192, 256, 0, stream>>>(Wqk, Wqkt, 1024, 2048);
  transpose_f32_f16<<<4096, 256, 0, stream>>>(Wv, Wvt, 1024, 1024);
  transpose_f32_f16<<<4096, 256, 0, stream>>>(Wo, Wot, 1024, 1024);

  dim3 gqk(64, 16);
  gemm16<<<gqk, 256, 0, stream>>>(Xf, Wqkt, bqk, qkb, 4096, 2048, 1024);
  dim3 gv(64, 8);
  gemm16<<<gv, 256, 0, stream>>>(Xf, Wvt, nullptr, vb, 4096, 1024, 1024);

  dim3 gnr(TT, BB);
  normrope<<<gnr, 256, 0, stream>>>(qkb, qw, kw, fc, fs, Q16, K16);
  transposeV<<<16384, 256, 0, stream>>>(vb, Vt16);

  dim3 ga(TT / 64, HH, BB);
  attn_fwd<<<ga, 128, 0, stream>>>(Q16, K16, Vt16, A16);

  gemm16<<<gv, 256, 0, stream>>>(A16, Wot, nullptr, out, 4096, 1024, 1024);
}
